// MultiHeadAttention_72249939853365
// MI455X (gfx1250) — matrix operand order repaired and checked by reading, not compiled, not run on hardware
//
#include <hip/hip_runtime.h>

// B=2, S=2048, D=1024, H=16, DH=64.  M = B*S = 4096.
// f16 compute via V_WMMA_F32_16X16X32_F16, f32 accumulate.

typedef _Float16 h4  __attribute__((ext_vector_type(4)));
typedef _Float16 h8  __attribute__((ext_vector_type(8)));
typedef _Float16 v16h __attribute__((ext_vector_type(16)));
typedef float    v8f  __attribute__((ext_vector_type(8)));

#define WMMA16(a, b, c) \
  __builtin_amdgcn_wmma_f32_16x16x32_f16(false, (a), false, (b), (short)0, (c), false, false)

static __device__ __forceinline__ v16h cat16(h8 lo, h8 hi) {
  return __builtin_shufflevector(lo, hi, 0,1,2,3,4,5,6,7,8,9,10,11,12,13,14,15);
}

static __device__ __forceinline__ float redmax16(float v) {
  v = fmaxf(v, __shfl_xor(v, 1, 32));
  v = fmaxf(v, __shfl_xor(v, 2, 32));
  v = fmaxf(v, __shfl_xor(v, 4, 32));
  v = fmaxf(v, __shfl_xor(v, 8, 32));
  return v;
}
static __device__ __forceinline__ float redsum16(float v) {
  v += __shfl_xor(v, 1, 32);
  v += __shfl_xor(v, 2, 32);
  v += __shfl_xor(v, 4, 32);
  v += __shfl_xor(v, 8, 32);
  return v;
}

// ---------------------------------------------------------------- convert
__global__ __launch_bounds__(256) void cvt_f32_to_f16(
    const float* __restrict__ src, _Float16* __restrict__ dst, int n4) {
  int i = blockIdx.x * blockDim.x + threadIdx.x;
  if (i < n4) {
    float4 v = ((const float4*)src)[i];
    h4 o;
    o.x = (_Float16)v.x; o.y = (_Float16)v.y;
    o.z = (_Float16)v.z; o.w = (_Float16)v.w;
    ((h4*)dst)[i] = o;
  }
}

// ---------------------------------------------------------------- QKV GEMM
// Out[m,n] = sum_k X[m,k] * W[n,k]   (NT GEMM), M=4096, N=K=1024.
// Block tile 128x128, 8 waves each doing 32x64 (2x4 WMMA tiles), K-step 32.
// z=0 -> Q row-major, z=1 -> K row-major, z=2 -> V transposed (B,D,S).
__global__ __launch_bounds__(256) void gemm_qkv_kernel(
    const _Float16* __restrict__ X,
    const _Float16* __restrict__ Wqh, const _Float16* __restrict__ Wkh,
    const _Float16* __restrict__ Wvh,
    _Float16* __restrict__ Q, _Float16* __restrict__ K, _Float16* __restrict__ Vt) {
  __shared__ _Float16 As[128][40];  // [m][k], 8-half pad -> conflict-free frags
  __shared__ _Float16 Bs[128][40];  // [n][k]

  const int z = blockIdx.z;
  const _Float16* __restrict__ W = (z == 0) ? Wqh : (z == 1) ? Wkh : Wvh;
  const int m0 = blockIdx.x * 128;
  const int n0 = blockIdx.y * 128;
  const int t = threadIdx.x;
  const int wid = t >> 5, lane = t & 31, lrow = lane & 15, hi = lane >> 4;
  const int wm = (wid & 3) * 32;   // 4 waves along M
  const int wn = (wid >> 2) * 64;  // 2 waves along N
  const int grow = t >> 1, gchunk = (t & 1) * 16;

  v8f acc[2][4] = {};

  const _Float16* pa = X + (size_t)(m0 + grow) * 1024 + gchunk;
  const _Float16* pb = W + (size_t)(n0 + grow) * 1024 + gchunk;

  for (int k0 = 0; k0 < 1024; k0 += 32) {
    float4 a0 = *(const float4*)(pa + k0);
    float4 a1 = *(const float4*)(pa + k0 + 8);
    float4 b0 = *(const float4*)(pb + k0);
    float4 b1 = *(const float4*)(pb + k0 + 8);
    __syncthreads();
    *(float4*)&As[grow][gchunk]     = a0;
    *(float4*)&As[grow][gchunk + 8] = a1;
    *(float4*)&Bs[grow][gchunk]     = b0;
    *(float4*)&Bs[grow][gchunk + 8] = b1;
    __syncthreads();

    v16h af[2], bf[4];
#pragma unroll
    for (int im = 0; im < 2; ++im) {
      const _Float16* ra = &As[wm + 16 * im + lrow][0];
      // A layout: lane row = lane&15; K chunks at 8*hi and 8*hi+16
      af[im] = cat16(*(const h8*)(ra + 8 * hi), *(const h8*)(ra + 8 * hi + 16));
    }
#pragma unroll
    for (int in = 0; in < 4; ++in) {
      const _Float16* rb = &Bs[wn + 16 * in + lrow][0];
      // B layout: lane col = lane&15; K chunks at 8*hi and 8*hi+16
      bf[in] = cat16(*(const h8*)(rb + 8 * hi), *(const h8*)(rb + 8 * hi + 16));
    }
#pragma unroll
    for (int im = 0; im < 2; ++im)
#pragma unroll
      for (int in = 0; in < 4; ++in)
        acc[im][in] = WMMA16(af[im], bf[in], acc[im][in]);
  }

  if (z < 2) {
    _Float16* O = (z == 0) ? Q : K;
#pragma unroll
    for (int im = 0; im < 2; ++im)
#pragma unroll
      for (int in = 0; in < 4; ++in)
#pragma unroll
        for (int r = 0; r < 8; ++r) {
          int m = m0 + wm + 16 * im + r + 8 * hi;  // C layout: M = r + 8*hi
          int n = n0 + wn + 16 * in + lrow;        //           N = lane&15
          O[(size_t)m * 1024 + n] = (_Float16)acc[im][in][r];
        }
  } else {
    // V transposed: Vt[(b*1024 + n)*2048 + s]; r=0..7 -> contiguous s
#pragma unroll
    for (int im = 0; im < 2; ++im)
#pragma unroll
      for (int in = 0; in < 4; ++in) {
        int mb = m0 + wm + 16 * im + 8 * hi;
        int bb = mb >> 11;
        int sb = mb & 2047;
        int n = n0 + wn + 16 * in + lrow;
        h8 pk;
#pragma unroll
        for (int r = 0; r < 8; ++r) pk[r] = (_Float16)acc[im][in][r];
        *(h8*)(Vt + ((size_t)(bb * 1024 + n)) * 2048 + sb) = pk;
      }
  }
}

// ---------------------------------------------------------------- flash attn
// One wave per 16-query tile per (b,h). Online-softmax causal attention.
__global__ __launch_bounds__(32) void flash_kernel(
    const _Float16* __restrict__ Qh, const _Float16* __restrict__ Kh,
    const _Float16* __restrict__ Vt, _Float16* __restrict__ Ctx) {
  __shared__ _Float16 Ps[16][40];  // P tile bounce buffer (C-layout -> A-layout)
  const int lane = threadIdx.x, lrow = lane & 15, hi = lane >> 4;
  const int q0 = blockIdx.x * 16, h = blockIdx.y, b = blockIdx.z;
  const size_t headoff = (size_t)b * 2048 * 1024 + h * 64;
  const float NEGINF = -3.0e38f;

  // Q A-fragments: 16 rows x 64 dh -> two 16x32 frags
  const _Float16* qp = Qh + headoff + (size_t)(q0 + lrow) * 1024;
  v16h qf[2];
#pragma unroll
  for (int f = 0; f < 2; ++f)
    qf[f] = cat16(*(const h8*)(qp + f * 32 + 8 * hi),
                  *(const h8*)(qp + f * 32 + 8 * hi + 16));

  v8f acc[4] = {};
  float mi[8], li[8];
#pragma unroll
  for (int r = 0; r < 8; ++r) { mi[r] = NEGINF; li[r] = 0.0f; }

  for (int j0 = 0; j0 <= q0; j0 += 32) {
#pragma unroll
    for (int sub = 0; sub < 2; ++sub) {
      const int kb = j0 + sub * 16;
      const int key = kb + lrow;
      const int krow = key < 2047 ? key : 2047;  // clamp OOB (masked anyway)
      const _Float16* kp = Kh + headoff + (size_t)krow * 1024;
      // K^T B-fragments: N = key (lane&15), K-dim = dh contiguous
      v16h kf0 = cat16(*(const h8*)(kp + 8 * hi), *(const h8*)(kp + 8 * hi + 16));
      v16h kf1 = cat16(*(const h8*)(kp + 32 + 8 * hi), *(const h8*)(kp + 32 + 8 * hi + 16));
      v8f s = {};
      s = WMMA16(qf[0], kf0, s);
      s = WMMA16(qf[1], kf1, s);
#pragma unroll
      for (int r = 0; r < 8; ++r) {
        int qrow = q0 + r + 8 * hi;
        float sv = (float)s[r] * 0.125f;  // scale = DH^-0.5
        if (key > qrow) sv = NEGINF;      // causal mask
        float rmax = redmax16(sv);
        float mnew = fmaxf(mi[r], rmax);
        float corr = __expf(mi[r] - mnew);
        float p = __expf(sv - mnew);
        float rsum = redsum16(p);
        li[r] = li[r] * corr + rsum;
        mi[r] = mnew;
#pragma unroll
        for (int tt = 0; tt < 4; ++tt) acc[tt][r] *= corr;
        Ps[r + 8 * hi][sub * 16 + lrow] = (_Float16)p;
      }
    }
    asm volatile("s_wait_dscnt 0" ::: "memory");  // same-wave LDS RAW
    // P as A-fragment (16 rows x 32 keys)
    const _Float16* pp = &Ps[lrow][0];
    v16h pf = cat16(*(const h8*)(pp + 16 * hi), *(const h8*)(pp + 16 * hi + 8));
    // V B-fragments from transposed V: N = dh, K-dim = key contiguous
    int ko = j0 + 16 * hi;
    if (ko > 2032) ko = 2032;  // clamp OOB (those P columns are exactly 0)
#pragma unroll
    for (int tt = 0; tt < 4; ++tt) {
      const _Float16* vp =
          Vt + ((size_t)(b * 1024 + h * 64 + tt * 16 + lrow)) * 2048 + ko;
      v16h vf = cat16(*(const h8*)vp, *(const h8*)(vp + 8));
      acc[tt] = WMMA16(pf, vf, acc[tt]);
    }
  }

#pragma unroll
  for (int tt = 0; tt < 4; ++tt)
#pragma unroll
    for (int r = 0; r < 8; ++r) {
      int q = q0 + r + 8 * hi;
      Ctx[headoff + (size_t)q * 1024 + tt * 16 + lrow] =
          (_Float16)(acc[tt][r] / li[r]);
    }
}

// ---------------------------------------------------------------- out GEMM
// out[m,n] = sum_k Ctx[m,k] * Wo[n,k] + bo[n], f32 output.
__global__ __launch_bounds__(256) void gemm_out_kernel(
    const _Float16* __restrict__ X, const _Float16* __restrict__ Wh,
    const float* __restrict__ bias, float* __restrict__ out) {
  __shared__ _Float16 As[128][40];
  __shared__ _Float16 Bs[128][40];
  const int m0 = blockIdx.x * 128;
  const int n0 = blockIdx.y * 128;
  const int t = threadIdx.x;
  const int wid = t >> 5, lane = t & 31, lrow = lane & 15, hi = lane >> 4;
  const int wm = (wid & 3) * 32;
  const int wn = (wid >> 2) * 64;
  const int grow = t >> 1, gchunk = (t & 1) * 16;

  v8f acc[2][4] = {};
  const _Float16* pa = X + (size_t)(m0 + grow) * 1024 + gchunk;
  const _Float16* pb = Wh + (size_t)(n0 + grow) * 1024 + gchunk;

  for (int k0 = 0; k0 < 1024; k0 += 32) {
    float4 a0 = *(const float4*)(pa + k0);
    float4 a1 = *(const float4*)(pa + k0 + 8);
    float4 b0 = *(const float4*)(pb + k0);
    float4 b1 = *(const float4*)(pb + k0 + 8);
    __syncthreads();
    *(float4*)&As[grow][gchunk]     = a0;
    *(float4*)&As[grow][gchunk + 8] = a1;
    *(float4*)&Bs[grow][gchunk]     = b0;
    *(float4*)&Bs[grow][gchunk + 8] = b1;
    __syncthreads();

    v16h af[2], bf[4];
#pragma unroll
    for (int im = 0; im < 2; ++im) {
      const _Float16* ra = &As[wm + 16 * im + lrow][0];
      af[im] = cat16(*(const h8*)(ra + 8 * hi), *(const h8*)(ra + 8 * hi + 16));
    }
#pragma unroll
    for (int in = 0; in < 4; ++in) {
      const _Float16* rb = &Bs[wn + 16 * in + lrow][0];
      bf[in] = cat16(*(const h8*)(rb + 8 * hi), *(const h8*)(rb + 8 * hi + 16));
    }
#pragma unroll
    for (int im = 0; im < 2; ++im)
#pragma unroll
      for (int in = 0; in < 4; ++in)
        acc[im][in] = WMMA16(af[im], bf[in], acc[im][in]);
  }

#pragma unroll
  for (int im = 0; im < 2; ++im)
#pragma unroll
    for (int in = 0; in < 4; ++in) {
      int n = n0 + wn + 16 * in + lrow;
      float bv = bias[n];
#pragma unroll
      for (int r = 0; r < 8; ++r) {
        int m = m0 + wm + 16 * im + r + 8 * hi;
        out[(size_t)m * 1024 + n] = acc[im][in][r] + bv;
      }
    }
}

// ---------------------------------------------------------------- launch
extern "C" void kernel_launch(void* const* d_in, const int* in_sizes, int n_in,
                              void* d_out, int out_size, void* d_ws, size_t ws_size,
                              hipStream_t stream) {
  (void)in_sizes; (void)n_in; (void)out_size; (void)ws_size;
  const float* x  = (const float*)d_in[0];
  const float* Wq = (const float*)d_in[1];
  const float* Wk = (const float*)d_in[2];
  const float* Wv = (const float*)d_in[3];
  const float* Wo = (const float*)d_in[4];
  const float* bo = (const float*)d_in[5];
  float* out = (float*)d_out;

  const size_t MD = (size_t)4096 * 1024;  // x / Q / K / V / ctx elements
  const size_t DD = (size_t)1024 * 1024;  // weight elements

  _Float16* p = (_Float16*)d_ws;
  _Float16* xh  = p; p += MD;
  _Float16* Wqh = p; p += DD;
  _Float16* Wkh = p; p += DD;
  _Float16* Wvh = p; p += DD;
  _Float16* Woh = p; p += DD;
  _Float16* Qh  = p; p += MD;
  _Float16* Kh  = p; p += MD;
  _Float16* Vt  = p; p += MD;   // transposed: (B, D, S)
  _Float16* Ctx = p; p += MD;

  {
    int n4 = (int)(MD / 4);
    cvt_f32_to_f16<<<(n4 + 255) / 256, 256, 0, stream>>>(x, xh, n4);
    n4 = (int)(DD / 4);
    cvt_f32_to_f16<<<(n4 + 255) / 256, 256, 0, stream>>>(Wq, Wqh, n4);
    cvt_f32_to_f16<<<(n4 + 255) / 256, 256, 0, stream>>>(Wk, Wkh, n4);
    cvt_f32_to_f16<<<(n4 + 255) / 256, 256, 0, stream>>>(Wv, Wvh, n4);
    cvt_f32_to_f16<<<(n4 + 255) / 256, 256, 0, stream>>>(Wo, Woh, n4);
  }

  gemm_qkv_kernel<<<dim3(32, 8, 3), 256, 0, stream>>>(xh, Wqh, Wkh, Wvh, Qh, Kh, Vt);
  flash_kernel<<<dim3(128, 16, 2), 32, 0, stream>>>(Qh, Kh, Vt, Ctx);
  gemm_out_kernel<<<dim3(32, 8), 256, 0, stream>>>(Ctx, Woh, bo, out);
}
